// VRNN_1829656068682
// MI455X (gfx1250) — matrix operand order repaired and checked by reading, not compiled, not run on hardware
//
#include <hip/hip_runtime.h>
#include <hip/hip_bf16.h>
#include <math.h>

typedef __attribute__((ext_vector_type(16))) __bf16 v16bf;
typedef __attribute__((ext_vector_type(8)))  float  v8f;
typedef __attribute__((ext_vector_type(4)))  unsigned int u32x4;

constexpr int NB  = 64;    // batch
constexpr int NT  = 256;   // time steps
constexpr int NWG = 64;    // persistent recurrence workgroups
constexpr int LDS_BYTES = 204800;   // 64KB small slice + 128KB lstm slice + 8KB scratch

union Frag { u32x4 q[2]; v16bf v; };

// ---------- WMMA fragment loaders (ISA 7.12.2 layouts, wave32) ----------

// A: 16x32 bf16, row-major: lanes 0-15 row M=lane hold K 0-7 & 16-23,
// lanes 16-31 row M=lane-16 hold K 8-15 & 24-31. Two b128 loads per lane.
__device__ __forceinline__ v16bf load_a_frag(const __bf16* base, int lda, int k0) {
  const int lane = threadIdx.x & 31;
  const int r    = lane & 15;
  const int koff = k0 + ((lane < 16) ? 0 : 8);
  const __bf16* p = base + r * lda + koff;
  Frag f;
  f.q[0] = *(const u32x4*)(p);
  f.q[1] = *(const u32x4*)(p + 16);
  return f.v;
}

// B: 32x16 bf16 from W^T [N][K] row-major: lane n holds column n,
// K 0-7 & 16-23 (lanes 0-15), K 8-15 & 24-31 (lanes 16-31).
__device__ __forceinline__ v16bf load_b_frag(const __bf16* wt, int ldw, int k0) {
  const int lane = threadIdx.x & 31;
  const int n    = lane & 15;
  const int kb   = k0 + ((lane < 16) ? 0 : 8);
  const __bf16* p = wt + n * ldw + kb;
  Frag f;
  f.q[0] = *(const u32x4*)(p);
  f.q[1] = *(const u32x4*)(p + 16);
  return f.v;
}

__device__ __forceinline__ v8f wmma_bf16(v16bf a, v16bf b, v8f c) {
  return __builtin_amdgcn_wmma_f32_16x16x32_bf16(false, a, false, b, (short)0, c,
                                                 false, false);
}

// C = [A1 | A2] @ Wt^T accumulated into acc. Bases pre-offset to the tile.
__device__ __forceinline__ v8f gemm_tile(const __bf16* A1, int lda1, int K1,
                                         const __bf16* A2, int lda2, int K2,
                                         const __bf16* Wt, int ldw, v8f acc) {
  for (int k = 0; k < K1; k += 32)
    acc = wmma_bf16(load_a_frag(A1, lda1, k), load_b_frag(Wt, ldw, k), acc);
  for (int k = 0; k < K2; k += 32)
    acc = wmma_bf16(load_a_frag(A2, lda2, k), load_b_frag(Wt, ldw, K1 + k), acc);
  return acc;
}

__device__ __forceinline__ float sigmoidf_(float x) { return 1.f / (1.f + __expf(-x)); }
__device__ __forceinline__ float softplusf_(float x) {
  return (x > 20.f) ? x : log1pf(__expf(x));
}

// C/D f32 16x16: VGPR r -> (M = r + 8*(lane>=16), N = lane&15)
__device__ __forceinline__ void store_tile_bf16(v8f acc, const float* bias,
                                                int rowbase, int colbase, int act,
                                                __bf16* out, int ldo, __bf16* out2) {
  const int lane = threadIdx.x & 31;
  const int nloc = lane & 15;
  const int mb   = (lane < 16) ? 0 : 8;
  const int col  = colbase + nloc;
  const float bv = bias ? bias[col] : 0.f;
#pragma unroll
  for (int r = 0; r < 8; ++r) {
    float v = acc[r] + bv;
    if (act == 1) v = fmaxf(v, 0.f);
    const int idx = (rowbase + mb + r) * ldo + col;
    out[idx] = (__bf16)v;
    if (out2) out2[idx] = (__bf16)v;
  }
}

// block-wide global -> LDS copy, 16B per thread per iteration
__device__ __forceinline__ void cp2lds(__bf16* dst, const __bf16* src, int elems) {
  u32x4* d = (u32x4*)dst;
  const u32x4* s = (const u32x4*)src;
  const int n = elems >> 3;
  for (int i = threadIdx.x; i < n; i += 256) d[i] = s[i];
}

// ---------------------- utility kernels ----------------------

__global__ void zero_u32(unsigned* p, int n) {
  int i = blockIdx.x * blockDim.x + threadIdx.x;
  if (i < n) p[i] = 0u;
}

// fp32 [K][N] -> bf16 [N][K] (W^T, out-feature-major)
__global__ void convert_wt(const float* src, __bf16* dst, int K, int N) {
  int i = blockIdx.x * blockDim.x + threadIdx.x;
  if (i >= K * N) return;
  int n = i % N, k = i / N;
  dst[n * K + k] = (__bf16)src[k * N + n];
}

// x [B][T][256] fp32 -> xb [(t*NB+b)*256 + d] bf16
__global__ void convert_x(const float* x, __bf16* xb) {
  int i = blockIdx.x * blockDim.x + threadIdx.x;
  if (i >= NB * NT * 256) return;
  int d = i & 255;
  int t = (i >> 8) & 255;
  int b = i >> 16;
  xb[((t << 6) + b) * 256 + d] = (__bf16)x[i];
}

// ---------------------- generic parallel GEMM ----------------------
// One wave = one 32x16 tile (2 M-subtiles sharing each B fragment).
__global__ void __launch_bounds__(256)
gemm_act(int M, int N,
         const __bf16* A1, int lda1, int K1,
         const __bf16* A2, int lda2, int K2,
         const __bf16* Wt, int ldw,
         const float* bias, int act,          // 0 none, 1 relu, 2 softplus+1e-4
         float* outF, int remap,              // remap: row=t*NB+b -> (b*NT+t)
         __bf16* outB) {
  const int wave = blockIdx.x * (blockDim.x >> 5) + (threadIdx.x >> 5);
  const int ntn  = N >> 4;
  const int tot  = (M >> 5) * ntn;
  if (wave >= tot) return;
  const int tm2 = wave / ntn, tn = wave % ntn;
  const int row0 = tm2 * 32;

  const __bf16* wt = Wt + (tn * 16) * ldw;
  v8f acc0 = {}, acc1 = {};
  for (int k = 0; k < K1; k += 32) {
    v16bf b = load_b_frag(wt, ldw, k);
    acc0 = wmma_bf16(load_a_frag(A1 + row0 * lda1, lda1, k), b, acc0);
    acc1 = wmma_bf16(load_a_frag(A1 + (row0 + 16) * lda1, lda1, k), b, acc1);
  }
  for (int k = 0; k < K2; k += 32) {
    v16bf b = load_b_frag(wt, ldw, K1 + k);
    acc0 = wmma_bf16(load_a_frag(A2 + row0 * lda2, lda2, k), b, acc0);
    acc1 = wmma_bf16(load_a_frag(A2 + (row0 + 16) * lda2, lda2, k), b, acc1);
  }

  const int lane = threadIdx.x & 31;
  const int nloc = lane & 15;
  const int mb   = (lane < 16) ? 0 : 8;
  const int col  = tn * 16 + nloc;
  const float bv = bias ? bias[col] : 0.f;
#pragma unroll
  for (int half = 0; half < 2; ++half) {
    v8f acc = half ? acc1 : acc0;
#pragma unroll
    for (int r = 0; r < 8; ++r) {
      const int row = row0 + half * 16 + mb + r;
      float v = acc[r] + bv;
      if (act == 1) v = fmaxf(v, 0.f);
      else if (act == 2) v = softplusf_(v) + 1e-4f;
      if (outF) {
        int idx;
        if (remap) { int t = row >> 6, b = row & 63; idx = (b * NT + t) * N + col; }
        else idx = row * N + col;
        outF[idx] = v;
      }
      if (outB) outB[row * N + col] = (__bf16)v;
    }
  }
}

// ---------------------- grid-wide barrier ----------------------
__device__ __forceinline__ void gsync(unsigned* sync) {
  __syncthreads();
  if (threadIdx.x == 0) {
    __threadfence();
    volatile unsigned* vgen = sync + 1;
    const unsigned gen = *vgen;
    const unsigned arrived = atomicAdd(sync, 1u);
    if (arrived == NWG - 1) {
      *(volatile unsigned*)sync = 0u;
      __threadfence();
      atomicAdd(sync + 1, 1u);
    } else {
      while (*vgen == gen) __builtin_amdgcn_s_sleep(1);
    }
    __threadfence();
  }
  __syncthreads();
}

// ---------------------- persistent recurrence ----------------------
// LDS (dynamic, 200KB): [0..32767] small-stage W slice, [32768..98303] LSTM
// slice (Wx g0..g3 then Wh g0..g3, 8192 elems each), then 8KB f32 scratch.
__global__ void __launch_bounds__(256)
vrnn_recur(const __bf16* xfeat_all,          // [T][64][256]
           const float* eps,                 // [B][T][256]
           const __bf16* WencHt, const float* enc_bh,
           const __bf16* WencMt, const float* enc_bm,
           const __bf16* WencSt, const float* enc_bs,
           const __bf16* Wpz1t, const float* pz_b1,
           const __bf16* Wpz2t, const float* pz_b2,
           const __bf16* Wl1xt, const __bf16* Wl1ht, const float* l1b,
           const __bf16* Wl2xt, const __bf16* Wl2ht, const float* l2b,
           __bf16* henc, __bf16* zbuf, __bf16* zf1, __bf16* zfeat,
           __bf16* zfeat_all,                // [T][64][256]
           __bf16* h1buf, __bf16* h2buf,     // [2][64][512] each
           float* c1, float* c2,             // [64][512] each
           __bf16* hprev_all,                // [T][64][512]
           float* out_meanz, float* out_scalez,
           unsigned* sync) {
  extern __shared__ char ldsraw[];
  __bf16* ldsSmall = (__bf16*)ldsraw;            // 32768 elems
  __bf16* ldsLstm  = ldsSmall + 32768;           // 65536 elems
  float*  ldsScr   = (float*)(ldsSmall + 98304); // 2048 floats

  const int bk   = blockIdx.x;
  const int w    = threadIdx.x >> 5;
  const int lane = threadIdx.x & 31;

  // ---- one-time weight preload into LDS (slices constant over t) ----
  if (bk < 16) {
    cp2lds(ldsSmall, WencHt + bk * 32 * 768, 32 * 768);
  } else if (bk < 24) {
    cp2lds(ldsSmall, WencMt + (bk - 16) * 32 * 512, 32 * 512);
    cp2lds(ldsSmall + 16384, WencSt + (bk - 16) * 32 * 512, 32 * 512);
  } else if (bk < 32) {
    cp2lds(ldsSmall, Wpz1t + (bk - 24) * 32 * 256, 32 * 256);
  } else if (bk < 40) {
    cp2lds(ldsSmall, Wpz2t + (bk - 32) * 32 * 256, 32 * 256);
  }
  if (bk < 32) {
#pragma unroll
    for (int g = 0; g < 4; ++g) {
      cp2lds(ldsLstm + g * 8192,         Wl1xt + (g * 512 + bk * 16) * 512, 8192);
      cp2lds(ldsLstm + 32768 + g * 8192, Wl1ht + (g * 512 + bk * 16) * 512, 8192);
    }
  } else {
    const int tn = bk - 32;
#pragma unroll
    for (int g = 0; g < 4; ++g) {
      cp2lds(ldsLstm + g * 8192,         Wl2xt + (g * 512 + tn * 16) * 512, 8192);
      cp2lds(ldsLstm + 32768 + g * 8192, Wl2ht + (g * 512 + tn * 16) * 512, 8192);
    }
  }
  __syncthreads();

  for (int t = 0; t < NT; ++t) {
    const int p = t & 1;
    const __bf16* h1  = h1buf + p * NB * 512;
    __bf16*       h1n = h1buf + (1 - p) * NB * 512;
    const __bf16* h2  = h2buf + p * NB * 512;
    __bf16*       h2n = h2buf + (1 - p) * NB * 512;
    const __bf16* xf  = xfeat_all + t * NB * 256;

    // ---- S0: henc = relu([xf|h2] @ enc_Wh) ; record h_prev ----
    if (bk < 16) {
      const int tnl = w >> 2, tm = w & 3;
      const int tn  = bk * 2 + tnl;
      v8f acc = {};
      acc = gemm_tile(xf + tm * 16 * 256, 256, 256,
                      h2 + tm * 16 * 512, 512, 512,
                      ldsSmall + tnl * 16 * 768, 768, acc);
      store_tile_bf16(acc, enc_bh, tm * 16, tn * 16, 1, henc, 512, nullptr);
    } else if (bk >= 48 && bk < 56) {
      const int idx = ((bk - 48) * 8 + w) * 32 + lane;     // 2048 threads
      const u32x4* s = (const u32x4*)h2;
      u32x4* d = (u32x4*)(hprev_all + t * NB * 512);
      d[idx * 2]     = s[idx * 2];
      d[idx * 2 + 1] = s[idx * 2 + 1];
    }
    gsync(sync);

    // ---- S1: mean_z, scale_z, z = mean + scale*eps ----
    if (bk >= 16 && bk < 24) {
      const int tnl = w >> 2, tm = w & 3;
      const int tn  = (bk - 16) * 2 + tnl;
      v8f am = {}, as = {};
      am = gemm_tile(henc + tm * 16 * 512, 512, 512, nullptr, 0, 0,
                     ldsSmall + tnl * 16 * 512, 512, am);
      as = gemm_tile(henc + tm * 16 * 512, 512, 512, nullptr, 0, 0,
                     ldsSmall + 16384 + tnl * 16 * 512, 512, as);
      const int nloc = lane & 15, mb = (lane < 16) ? 0 : 8;
      const int col  = tn * 16 + nloc;
      const float bm = enc_bm[col], bs = enc_bs[col];
#pragma unroll
      for (int r = 0; r < 8; ++r) {
        const int b  = tm * 16 + mb + r;
        const float mz = am[r] + bm;
        const float sz = softplusf_(as[r] + bs) + 1e-4f;
        const float e  = eps[(b * NT + t) * 256 + col];
        out_meanz[(b * NT + t) * 256 + col]  = mz;
        out_scalez[(b * NT + t) * 256 + col] = sz;
        zbuf[b * 256 + col] = (__bf16)(mz + sz * e);
      }
    }
    gsync(sync);

    // ---- S2: phi_z layer 1 (relu) ----
    if (bk >= 24 && bk < 32) {
      const int tnl = w >> 2, tm = w & 3;
      const int tn  = (bk - 24) * 2 + tnl;
      v8f acc = {};
      acc = gemm_tile(zbuf + tm * 16 * 256, 256, 256, nullptr, 0, 0,
                      ldsSmall + tnl * 16 * 256, 256, acc);
      store_tile_bf16(acc, pz_b1, tm * 16, tn * 16, 1, zf1, 256, nullptr);
    }
    gsync(sync);

    // ---- S3: phi_z layer 2 -> zfeat + zfeat_all[t] ----
    if (bk >= 32 && bk < 40) {
      const int tnl = w >> 2, tm = w & 3;
      const int tn  = (bk - 32) * 2 + tnl;
      v8f acc = {};
      acc = gemm_tile(zf1 + tm * 16 * 256, 256, 256, nullptr, 0, 0,
                      ldsSmall + tnl * 16 * 256, 256, acc);
      store_tile_bf16(acc, pz_b2, tm * 16, tn * 16, 0, zfeat, 256,
                      zfeat_all + t * NB * 256);
    }
    gsync(sync);

    // ---- S4: LSTM1, gates split across wave halves ----
    if (bk < 32) {
      const int tm = w & 3;
      const int gh = w >> 2;                  // 0: gates i,f   1: gates g,o
      const int ga_i = gh * 2, gb_i = gh * 2 + 1;
      const __bf16* aX = xf + tm * 16 * 256;
      const __bf16* aZ = zfeat + tm * 16 * 256;
      const __bf16* aH = h1 + tm * 16 * 512;
      v8f ga = {}, gb = {};
      ga = gemm_tile(aX, 256, 256, aZ, 256, 256, ldsLstm + ga_i * 8192, 512, ga);
      ga = gemm_tile(aH, 512, 512, nullptr, 0, 0,
                     ldsLstm + 32768 + ga_i * 8192, 512, ga);
      gb = gemm_tile(aX, 256, 256, aZ, 256, 256, ldsLstm + gb_i * 8192, 512, gb);
      gb = gemm_tile(aH, 512, 512, nullptr, 0, 0,
                     ldsLstm + 32768 + gb_i * 8192, 512, gb);
      if (gh == 1) {
#pragma unroll
        for (int r = 0; r < 8; ++r) {
          ldsScr[tm * 256 + r * 32 + lane]         = ga[r];   // gate g
          ldsScr[(4 + tm) * 256 + r * 32 + lane]   = gb[r];   // gate o
        }
      }
      __syncthreads();
      if (gh == 0) {
        const int nloc = lane & 15, mb = (lane < 16) ? 0 : 8;
        const int col  = bk * 16 + nloc;
        const float bi = l1b[col], bff = l1b[512 + col];
        const float bg = l1b[1024 + col], bo = l1b[1536 + col];
#pragma unroll
        for (int r = 0; r < 8; ++r) {
          const int b = tm * 16 + mb + r;
          const float g2 = ldsScr[tm * 256 + r * 32 + lane];
          const float g3 = ldsScr[(4 + tm) * 256 + r * 32 + lane];
          const float iv = sigmoidf_(ga[r] + bi);
          const float fv = sigmoidf_(gb[r] + bff);
          const float gv = tanhf(g2 + bg);
          const float ov = sigmoidf_(g3 + bo);
          const float cn = fv * c1[b * 512 + col] + iv * gv;
          c1[b * 512 + col] = cn;
          h1n[b * 512 + col] = (__bf16)(ov * tanhf(cn));
        }
      }
    }
    gsync(sync);

    // ---- S5: LSTM2, same scheme on blocks 32..63 ----
    if (bk >= 32) {
      const int tn = bk - 32;
      const int tm = w & 3;
      const int gh = w >> 2;
      const int ga_i = gh * 2, gb_i = gh * 2 + 1;
      const __bf16* aA = h1n + tm * 16 * 512;
      const __bf16* aB = h2 + tm * 16 * 512;
      v8f ga = {}, gb = {};
      ga = gemm_tile(aA, 512, 512, nullptr, 0, 0, ldsLstm + ga_i * 8192, 512, ga);
      ga = gemm_tile(aB, 512, 512, nullptr, 0, 0,
                     ldsLstm + 32768 + ga_i * 8192, 512, ga);
      gb = gemm_tile(aA, 512, 512, nullptr, 0, 0, ldsLstm + gb_i * 8192, 512, gb);
      gb = gemm_tile(aB, 512, 512, nullptr, 0, 0,
                     ldsLstm + 32768 + gb_i * 8192, 512, gb);
      if (gh == 1) {
#pragma unroll
        for (int r = 0; r < 8; ++r) {
          ldsScr[tm * 256 + r * 32 + lane]       = ga[r];
          ldsScr[(4 + tm) * 256 + r * 32 + lane] = gb[r];
        }
      }
      __syncthreads();
      if (gh == 0) {
        const int nloc = lane & 15, mb = (lane < 16) ? 0 : 8;
        const int col  = tn * 16 + nloc;
        const float bi = l2b[col], bff = l2b[512 + col];
        const float bg = l2b[1024 + col], bo = l2b[1536 + col];
#pragma unroll
        for (int r = 0; r < 8; ++r) {
          const int b = tm * 16 + mb + r;
          const float g2 = ldsScr[tm * 256 + r * 32 + lane];
          const float g3 = ldsScr[(4 + tm) * 256 + r * 32 + lane];
          const float iv = sigmoidf_(ga[r] + bi);
          const float fv = sigmoidf_(gb[r] + bff);
          const float gv = tanhf(g2 + bg);
          const float ov = sigmoidf_(g3 + bo);
          const float cn = fv * c2[b * 512 + col] + iv * gv;
          c2[b * 512 + col] = cn;
          h2n[b * 512 + col] = (__bf16)(ov * tanhf(cn));
        }
      }
    }
    gsync(sync);
  }
}

// ---------------------- host orchestration ----------------------
extern "C" void kernel_launch(void* const* d_in, const int* in_sizes, int n_in,
                              void* d_out, int out_size, void* d_ws, size_t ws_size,
                              hipStream_t stream) {
  (void)in_sizes; (void)n_in; (void)out_size; (void)ws_size;
  const float* x        = (const float*)d_in[0];
  const float* eps      = (const float*)d_in[1];
  const float* phix_W1  = (const float*)d_in[2];
  const float* phix_b1  = (const float*)d_in[3];
  const float* phix_W2  = (const float*)d_in[4];
  const float* phix_b2  = (const float*)d_in[5];
  const float* phiz_W1  = (const float*)d_in[6];
  const float* phiz_b1  = (const float*)d_in[7];
  const float* phiz_W2  = (const float*)d_in[8];
  const float* phiz_b2  = (const float*)d_in[9];
  const float* enc_Wh   = (const float*)d_in[10];
  const float* enc_bh   = (const float*)d_in[11];
  const float* enc_Wm   = (const float*)d_in[12];
  const float* enc_bm   = (const float*)d_in[13];
  const float* enc_Ws   = (const float*)d_in[14];
  const float* enc_bs   = (const float*)d_in[15];
  const float* prior_Wh = (const float*)d_in[16];
  const float* prior_bh = (const float*)d_in[17];
  const float* prior_Wm = (const float*)d_in[18];
  const float* prior_bm = (const float*)d_in[19];
  const float* prior_Ws = (const float*)d_in[20];
  const float* prior_bs = (const float*)d_in[21];
  const float* dec_Wh   = (const float*)d_in[22];
  const float* dec_bh   = (const float*)d_in[23];
  const float* dec_Wm   = (const float*)d_in[24];
  const float* dec_bm   = (const float*)d_in[25];
  const float* dec_Ws   = (const float*)d_in[26];
  const float* dec_bs   = (const float*)d_in[27];
  const float* l1_Wx    = (const float*)d_in[28];
  const float* l1_Wh    = (const float*)d_in[29];
  const float* l1_b     = (const float*)d_in[30];
  const float* l2_Wx    = (const float*)d_in[31];
  const float* l2_Wh    = (const float*)d_in[32];
  const float* l2_b     = (const float*)d_in[33];

  char* ws = (char*)d_ws;
  size_t off = 0;
  auto alloc = [&](size_t bytes) -> void* {
    off = (off + 255) & ~(size_t)255;
    void* p = ws + off;
    off += bytes;
    return p;
  };
  auto abf = [&](size_t e) -> __bf16* { return (__bf16*)alloc(e * 2); };

  __bf16* WphixW1t = abf(256 * 256);
  __bf16* WphixW2t = abf(256 * 256);
  __bf16* WphizW1t = abf(256 * 256);
  __bf16* WphizW2t = abf(256 * 256);
  __bf16* WencHt   = abf(512 * 768);
  __bf16* WencMt   = abf(256 * 512);
  __bf16* WencSt   = abf(256 * 512);
  __bf16* WprHt    = abf(512 * 512);
  __bf16* WprMt    = abf(256 * 512);
  __bf16* WprSt    = abf(256 * 512);
  __bf16* WdecHt   = abf(512 * 768);
  __bf16* WdecMt   = abf(256 * 512);
  __bf16* WdecSt   = abf(256 * 512);
  __bf16* Wl1xt    = abf(2048 * 512);
  __bf16* Wl1ht    = abf(2048 * 512);
  __bf16* Wl2xt    = abf(2048 * 512);
  __bf16* Wl2ht    = abf(2048 * 512);
  const size_t MR = (size_t)NB * NT;                 // 16384 rows
  __bf16* xb        = abf(MR * 256);
  __bf16* xh        = abf(MR * 256);
  __bf16* xfeat_all = abf(MR * 256);
  __bf16* zfeat_all = abf(MR * 256);
  __bf16* hprev_all = abf(MR * 512);
  __bf16* ph        = abf(MR * 512);
  __bf16* henc  = abf(NB * 512);
  __bf16* zbuf  = abf(NB * 256);
  __bf16* zf1   = abf(NB * 256);
  __bf16* zfeat = abf(NB * 256);
  __bf16* h1buf = abf(2 * NB * 512);
  __bf16* h2buf = abf(2 * NB * 512);
  float*  c1    = (float*)alloc(NB * 512 * 4);
  float*  c2    = (float*)alloc(NB * 512 * 4);
  unsigned* sync = (unsigned*)alloc(256);

  hipFuncSetAttribute((const void*)vrnn_recur,
                      hipFuncAttributeMaxDynamicSharedMemorySize, LDS_BYTES);

  auto Tw = [&](const float* src, __bf16* dst, int K, int N) {
    int tot = K * N;
    convert_wt<<<(tot + 255) / 256, 256, 0, stream>>>(src, dst, K, N);
  };
  Tw(phix_W1, WphixW1t, 256, 256);  Tw(phix_W2, WphixW2t, 256, 256);
  Tw(phiz_W1, WphizW1t, 256, 256);  Tw(phiz_W2, WphizW2t, 256, 256);
  Tw(enc_Wh, WencHt, 768, 512);
  Tw(enc_Wm, WencMt, 512, 256);     Tw(enc_Ws, WencSt, 512, 256);
  Tw(prior_Wh, WprHt, 512, 512);
  Tw(prior_Wm, WprMt, 512, 256);    Tw(prior_Ws, WprSt, 512, 256);
  Tw(dec_Wh, WdecHt, 768, 512);
  Tw(dec_Wm, WdecMt, 512, 256);     Tw(dec_Ws, WdecSt, 512, 256);
  Tw(l1_Wx, Wl1xt, 512, 2048);      Tw(l1_Wh, Wl1ht, 512, 2048);
  Tw(l2_Wx, Wl2xt, 512, 2048);      Tw(l2_Wh, Wl2ht, 512, 2048);

  convert_x<<<(NB * NT * 256 + 255) / 256, 256, 0, stream>>>(x, xb);

  zero_u32<<<(32768 + 255) / 256, 256, 0, stream>>>((unsigned*)h1buf, 32768);
  zero_u32<<<(32768 + 255) / 256, 256, 0, stream>>>((unsigned*)h2buf, 32768);
  zero_u32<<<(32768 + 255) / 256, 256, 0, stream>>>((unsigned*)c1, 32768);
  zero_u32<<<(32768 + 255) / 256, 256, 0, stream>>>((unsigned*)c2, 32768);
  zero_u32<<<1, 64, 0, stream>>>(sync, 64);

  auto G = [&](int M, int N, const __bf16* A1, int lda1, int K1,
               const __bf16* A2, int lda2, int K2,
               const __bf16* Wt, const float* bias, int act,
               float* outF, int remap, __bf16* outB) {
    int waves = (M / 32) * (N / 16);
    gemm_act<<<(waves + 7) / 8, 256, 0, stream>>>(M, N, A1, lda1, K1, A2, lda2, K2,
                                                  Wt, K1 + K2, bias, act, outF,
                                                  remap, outB);
  };

  // hoisted phi_x over all B*T rows
  G((int)MR, 256, xb, 256, 256, nullptr, 0, 0, WphixW1t, phix_b1, 1, nullptr, 0, xh);
  G((int)MR, 256, xh, 256, 256, nullptr, 0, 0, WphixW2t, phix_b2, 0, nullptr, 0,
    xfeat_all);

  float* out = (float*)d_out;
  const size_t OUTSZ = (size_t)NB * NT * 256;

  vrnn_recur<<<NWG, 256, LDS_BYTES, stream>>>(
      xfeat_all, eps,
      WencHt, enc_bh, WencMt, enc_bm, WencSt, enc_bs,
      WphizW1t, phiz_b1, WphizW2t, phiz_b2,
      Wl1xt, Wl1ht, l1_b, Wl2xt, Wl2ht, l2_b,
      henc, zbuf, zf1, zfeat, zfeat_all,
      h1buf, h2buf, c1, c2, hprev_all,
      out + 0 * OUTSZ, out + 1 * OUTSZ, sync);

  // deferred prior(h_prev): mean_0, scale_0
  G((int)MR, 512, hprev_all, 512, 512, nullptr, 0, 0, WprHt, prior_bh, 1,
    nullptr, 0, ph);
  G((int)MR, 256, ph, 512, 512, nullptr, 0, 0, WprMt, prior_bm, 0,
    out + 2 * OUTSZ, 1, nullptr);
  G((int)MR, 256, ph, 512, 512, nullptr, 0, 0, WprSt, prior_bs, 2,
    out + 3 * OUTSZ, 1, nullptr);

  // deferred dec([z_feat, h_prev]): mean_x, scale_x
  G((int)MR, 512, zfeat_all, 256, 256, hprev_all, 512, 512, WdecHt, dec_bh, 1,
    nullptr, 0, ph);
  G((int)MR, 256, ph, 512, 512, nullptr, 0, 0, WdecMt, dec_bm, 0,
    out + 4 * OUTSZ, 1, nullptr);
  G((int)MR, 256, ph, 512, 512, nullptr, 0, 0, WdecSt, dec_bs, 2,
    out + 5 * OUTSZ, 1, nullptr);
}
